// GraphCastProcessor_86303072846451
// MI455X (gfx1250) — matrix operand order repaired and checked by reading, not compiled, not run on hardware
//
#include <hip/hip_runtime.h>
#include <cstdint>

// ---------------------------------------------------------------------------
// GraphCast processor for MI455X (gfx1250): bf16 WMMA GEMMs, fp32 LN/residual,
// async global->LDS staging, deterministic CSR segment-sum.
// ---------------------------------------------------------------------------

#define N_NODES 4096
#define N_EDGES 16384
#define DFEAT   512
#define NUM_BLOCKS 16

#ifndef USE_ASYNC_LDS
#define USE_ASYNC_LDS 1     // gfx1250 GLOBAL_LOAD_ASYNC_TO_LDS_* path
#endif

typedef __attribute__((ext_vector_type(16))) __bf16 v16bf;
typedef __attribute__((ext_vector_type(8)))  float  v8f;

union FragU { uint4 u[2]; v16bf v; };

__device__ __forceinline__ v16bf mkfrag(uint4 a, uint4 b) {
    FragU f; f.u[0] = a; f.u[1] = b; return f.v;
}

__device__ __forceinline__ unsigned short f2bf(float f) {
    unsigned int x = __float_as_uint(f);
    unsigned int r = x + 0x7fffu + ((x >> 16) & 1u);   // round-to-nearest-even
    return (unsigned short)(r >> 16);
}

// 16B global -> LDS copy: async (ASYNCcnt, no VGPR landing zone) or fallback.
__device__ __forceinline__ void ld16_to_lds(unsigned lds_addr, const void* g) {
#if USE_ASYNC_LDS
    asm volatile("global_load_async_to_lds_b128 %0, %1, off"
                 :: "v"(lds_addr), "v"((unsigned long long)(uintptr_t)g)
                 : "memory");
#else
    *(uint4*)(uintptr_t)lds_addr = *(const uint4*)g;
#endif
}

__device__ __forceinline__ void pipeline_sync() {
#if USE_ASYNC_LDS
    asm volatile("s_wait_asynccnt 0x0" ::: "memory");
#endif
    __syncthreads();
}

// ---------------------------------------------------------------------------
// Weight pack: fp32 [nmat][K][512] -> bf16 WMMA-B tiles (tile = 256 u32,
// entry n*16+kp = pack(W[kt*32+2kp][nt*16+n], W[kt*32+2kp+1][nt*16+n])).
// Lane l reads its B fragment as 8 consecutive u32 at (l&15)*16 + (l>=16)*8.
// ---------------------------------------------------------------------------
__global__ void pack_weights_kernel(const float* __restrict__ src,
                                    unsigned int* __restrict__ dst,
                                    int K, int totalU32) {
    int idx = blockIdx.x * 256 + threadIdx.x;
    if (idx >= totalU32) return;
    int epm  = K * 256;
    int m    = idx / epm;
    int rem  = idx - m * epm;
    int t    = rem & 255;
    int tile = rem >> 8;
    int nt   = tile & 31;
    int kt   = tile >> 5;
    int n    = t >> 4;
    int kp   = t & 15;
    int k    = kt * 32 + kp * 2;
    int col  = nt * 16 + n;
    const float* s = src + (size_t)m * K * 512;
    float f0 = s[(size_t)k * 512 + col];
    float f1 = s[(size_t)(k + 1) * 512 + col];
    dst[idx] = (unsigned int)f2bf(f0) | ((unsigned int)f2bf(f1) << 16);
}

__global__ void copyconv_kernel(const float* __restrict__ src,
                                float* __restrict__ dstF,
                                unsigned short* __restrict__ dstB,
                                int count) {
    int i = blockIdx.x * 256 + threadIdx.x;
    if (i >= count) return;
    float v = src[i];
    if (dstF) dstF[i] = v;
    dstB[i] = f2bf(v);
}

__global__ void embed_h_kernel(const float* __restrict__ ef,
                               const float* __restrict__ w0,
                               const float* __restrict__ b0,
                               unsigned short* __restrict__ H) {
    int i = blockIdx.x * 256 + threadIdx.x;   // over E*512
    int j = i & 511;
    int e = i >> 9;
    float4 x = *(const float4*)(ef + (size_t)e * 4);
    float v = b0[j] + x.x * w0[j] + x.y * w0[512 + j]
                    + x.z * w0[1024 + j] + x.w * w0[1536 + j];
    float s = v / (1.f + __expf(-v));
    H[i] = f2bf(s);
}

// ------------------------- deterministic CSR build -------------------------
__global__ void degree_kernel(const int* __restrict__ dstI, int* __restrict__ deg) {
    int e = blockIdx.x * 256 + threadIdx.x;
    if (e < N_EDGES) atomicAdd(&deg[dstI[e]], 1);
}

__global__ void scan_offsets_kernel(const int* __restrict__ deg, int* __restrict__ offs) {
    __shared__ int part[1024];
    int t = threadIdx.x;
    int4 d = *(const int4*)(deg + t * 4);
    int sum4 = d.x + d.y + d.z + d.w;
    part[t] = sum4;
    __syncthreads();
    for (int off = 1; off < 1024; off <<= 1) {
        int v  = part[t];
        int vv = (t >= off) ? part[t - off] : 0;
        __syncthreads();
        part[t] = v + vv;
        __syncthreads();
    }
    int excl = part[t] - sum4;
    offs[t * 4]     = excl;
    offs[t * 4 + 1] = excl + d.x;
    offs[t * 4 + 2] = excl + d.x + d.y;
    offs[t * 4 + 3] = excl + d.x + d.y + d.z;
    if (t == 1023) offs[4096] = part[1023];
}

// Stable bucket fill: thread-per-node scan preserves ascending edge order.
__global__ void bucket_kernel(const int* __restrict__ dstI,
                              const int* __restrict__ offs,
                              int* __restrict__ elist) {
    int n = blockIdx.x * 256 + threadIdx.x;   // 4096 nodes
    int w = offs[n];
    for (int e = 0; e < N_EDGES; ++e)
        if (dstI[e] == n) elist[w++] = e;
}

// Deterministic segment-sum gather: agg_bf16[n] = bf16( sum_{e in CSR[n]} e_f32[e] )
__global__ void aggregate_gather_kernel(const float* __restrict__ E,
                                        const int* __restrict__ offs,
                                        const int* __restrict__ elist,
                                        unsigned short* __restrict__ aggB) {
    int n = blockIdx.x;
    int c = threadIdx.x * 2;
    int b0 = offs[n], b1 = offs[n + 1];
    float s0 = 0.f, s1 = 0.f;
    for (int i = b0; i < b1; ++i) {
        const float* row = E + (size_t)elist[i] * 512;
        s0 += row[c];
        s1 += row[c + 1];
    }
    aggB[(size_t)n * 512 + c]     = f2bf(s0);
    aggB[(size_t)n * 512 + c + 1] = f2bf(s1);
}

// ---------------------------------------------------------------------------
// GEMM #1: h = silu(A @ W + b), bf16 out.  WG = 256 thr (8 waves) computes
// 64 rows x 512 cols; wave (wm,wn) owns 32 rows x 128 cols = 16 C tiles, so
// each B fragment feeds two WMMAs.  A double-buffered in LDS via async copy.
//  MODE 0: row r = A0[r]                        (K=512)
//  MODE 1: row r = [n[r], agg[r]]               (K=1024)
//  MODE 2: row r = [e[r], n[src[r]], n[dst[r]]] (K=1536)
// ---------------------------------------------------------------------------
template<int K, int MODE>
__global__ __launch_bounds__(256) void gemm_silu_kernel(
    const unsigned short* __restrict__ A0,
    const unsigned short* __restrict__ A1,
    const int* __restrict__ srcI, const int* __restrict__ dstI,
    const unsigned int* __restrict__ Wt,
    const float* __restrict__ bias,
    unsigned short* __restrict__ Hout) {
    constexpr int KCH = K / 64;
    __shared__ unsigned int aL[2 * 2048];   // 2 x (64 rows x 64 bf16) = 16 KB

    const int tid  = threadIdx.x;
    const int lane = tid & 31;
    const int w    = tid >> 5;
    const int wm   = w >> 2, wn = w & 3;
    const int row0 = blockIdx.x * 64;

    // staging: thread owns 16B segment (seg) of rows srow and srow+32
    const int srow = tid >> 3;
    const int seg  = tid & 7;
    const int rgA  = row0 + srow;
    const int rgB  = rgA + 32;
    const unsigned short *q0a = A0 + (size_t)rgA * 512, *q0b = A0 + (size_t)rgB * 512;
    const unsigned short *q1a = nullptr, *q1b = nullptr, *q2a = nullptr, *q2b = nullptr;
    if (MODE == 2) {
        q1a = A1 + (size_t)srcI[rgA] * 512;  q1b = A1 + (size_t)srcI[rgB] * 512;
        q2a = A1 + (size_t)dstI[rgA] * 512;  q2b = A1 + (size_t)dstI[rgB] * 512;
    } else if (MODE == 1) {
        q1a = A1 + (size_t)rgA * 512;        q1b = A1 + (size_t)rgB * 512;
    }
    const unsigned ldsLo = (unsigned)(uintptr_t)&aL[srow * 32 + seg * 4];

    auto stage = [&](int c, int buf) {
        int k = c * 64 + seg * 8;
        const unsigned short *pa, *pb;
        if (MODE == 2) {
            pa = (k < 512) ? (q0a + k) : (k < 1024) ? (q1a + (k - 512)) : (q2a + (k - 1024));
            pb = (k < 512) ? (q0b + k) : (k < 1024) ? (q1b + (k - 512)) : (q2b + (k - 1024));
        } else if (MODE == 1) {
            pa = (k < 512) ? (q0a + k) : (q1a + (k - 512));
            pb = (k < 512) ? (q0b + k) : (q1b + (k - 512));
        } else {
            pa = q0a + k;  pb = q0b + k;
        }
        unsigned lo = ldsLo + buf * 8192;
        ld16_to_lds(lo,        pa);
        ld16_to_lds(lo + 4096, pb);
    };

    v8f acc0[8], acc1[8];
#pragma unroll
    for (int j = 0; j < 8; ++j) {
        acc0[j] = (v8f){0.f,0.f,0.f,0.f,0.f,0.f,0.f,0.f};
        acc1[j] = (v8f){0.f,0.f,0.f,0.f,0.f,0.f,0.f,0.f};
    }

    const int half = lane >> 4;
    const int l15  = lane & 15;
    const int boff = l15 * 16 + half * 8;

    stage(0, 0);
    pipeline_sync();
    for (int c = 0; c < KCH; ++c) {
        int buf = c & 1;
        if (c + 1 < KCH) stage(c + 1, buf ^ 1);
#pragma unroll
        for (int ks = 0; ks < 2; ++ks) {
            int k32 = c * 2 + ks;
            int ab  = buf * 2048 + (wm * 32 + l15) * 32 + ks * 16 + half * 8;
            v16bf a0 = mkfrag(*(const uint4*)&aL[ab],       *(const uint4*)&aL[ab + 4]);
            v16bf a1 = mkfrag(*(const uint4*)&aL[ab + 512], *(const uint4*)&aL[ab + 516]);
#pragma unroll
            for (int j = 0; j < 8; ++j) {
                const unsigned int* bt = Wt + ((size_t)k32 * 32 + wn * 8 + j) * 256 + boff;
                v16bf bf = mkfrag(*(const uint4*)bt, *(const uint4*)(bt + 4));
                acc0[j] = __builtin_amdgcn_wmma_f32_16x16x32_bf16(
                    false, a0, false, bf, (short)0, acc0[j], false, false);
                acc1[j] = __builtin_amdgcn_wmma_f32_16x16x32_bf16(
                    false, a1, false, bf, (short)0, acc1[j], false, false);
            }
        }
        pipeline_sync();
    }

    // epilogue: bias + SiLU -> bf16
#pragma unroll
    for (int j = 0; j < 8; ++j) {
        int col = wn * 128 + j * 16 + l15;
        float b = bias[col];
#pragma unroll
        for (int r = 0; r < 8; ++r) {
            int rl = wm * 32 + r + half * 8;
            float v0 = acc0[j][r] + b;
            float v1 = acc1[j][r] + b;
            Hout[(size_t)(row0 + rl) * 512 + col]      = f2bf(v0 / (1.f + __expf(-v0)));
            Hout[(size_t)(row0 + rl + 16) * 512 + col] = f2bf(v1 / (1.f + __expf(-v1)));
        }
    }
}

// ---------------------------------------------------------------------------
// GEMM #2 + LayerNorm (+ residual): out = base + LN(h@W + b)*g + be, K = 512.
// Same 64x512 tiling; row stats via shfl_xor + LDS float atomics.
// ---------------------------------------------------------------------------
template<bool HAS_BASE>
__global__ __launch_bounds__(256) void gemm_ln_kernel(
    const unsigned short* __restrict__ A,
    const unsigned int* __restrict__ Wt,
    const float* __restrict__ bias,
    const float* __restrict__ gam,
    const float* __restrict__ bet,
    float* __restrict__ OutF,
    unsigned short* __restrict__ OutB) {
    constexpr int KCH = 8;   // 512/64
    __shared__ unsigned int aL[2 * 2048];
    __shared__ float lsum[64], lsum2[64];

    const int tid  = threadIdx.x;
    const int lane = tid & 31;
    const int w    = tid >> 5;
    const int wm   = w >> 2, wn = w & 3;
    const int row0 = blockIdx.x * 64;

    if (tid < 64) { lsum[tid] = 0.f; lsum2[tid] = 0.f; }

    const int srow = tid >> 3;
    const int seg  = tid & 7;
    const unsigned short* q0a = A + (size_t)(row0 + srow) * 512;
    const unsigned short* q0b = q0a + (size_t)32 * 512;
    const unsigned ldsLo = (unsigned)(uintptr_t)&aL[srow * 32 + seg * 4];

    auto stage = [&](int c, int buf) {
        int k = c * 64 + seg * 8;
        unsigned lo = ldsLo + buf * 8192;
        ld16_to_lds(lo,        q0a + k);
        ld16_to_lds(lo + 4096, q0b + k);
    };

    v8f acc0[8], acc1[8];
#pragma unroll
    for (int j = 0; j < 8; ++j) {
        acc0[j] = (v8f){0.f,0.f,0.f,0.f,0.f,0.f,0.f,0.f};
        acc1[j] = (v8f){0.f,0.f,0.f,0.f,0.f,0.f,0.f,0.f};
    }

    const int half = lane >> 4;
    const int l15  = lane & 15;
    const int boff = l15 * 16 + half * 8;

    stage(0, 0);
    pipeline_sync();
    for (int c = 0; c < KCH; ++c) {
        int buf = c & 1;
        if (c + 1 < KCH) stage(c + 1, buf ^ 1);
#pragma unroll
        for (int ks = 0; ks < 2; ++ks) {
            int k32 = c * 2 + ks;
            int ab  = buf * 2048 + (wm * 32 + l15) * 32 + ks * 16 + half * 8;
            v16bf a0 = mkfrag(*(const uint4*)&aL[ab],       *(const uint4*)&aL[ab + 4]);
            v16bf a1 = mkfrag(*(const uint4*)&aL[ab + 512], *(const uint4*)&aL[ab + 516]);
#pragma unroll
            for (int j = 0; j < 8; ++j) {
                const unsigned int* bt = Wt + ((size_t)k32 * 32 + wn * 8 + j) * 256 + boff;
                v16bf bf = mkfrag(*(const uint4*)bt, *(const uint4*)(bt + 4));
                acc0[j] = __builtin_amdgcn_wmma_f32_16x16x32_bf16(
                    false, a0, false, bf, (short)0, acc0[j], false, false);
                acc1[j] = __builtin_amdgcn_wmma_f32_16x16x32_bf16(
                    false, a1, false, bf, (short)0, acc1[j], false, false);
            }
        }
        pipeline_sync();
    }

    // bias add
#pragma unroll
    for (int j = 0; j < 8; ++j) {
        float b = bias[wn * 128 + j * 16 + l15];
#pragma unroll
        for (int r = 0; r < 8; ++r) { acc0[j][r] += b; acc1[j][r] += b; }
    }
    // per-row stats over this wave's 128 cols -> LDS atomics
#pragma unroll
    for (int mt = 0; mt < 2; ++mt) {
#pragma unroll
        for (int r = 0; r < 8; ++r) {
            float s = 0.f, s2 = 0.f;
#pragma unroll
            for (int j = 0; j < 8; ++j) {
                float v = mt ? acc1[j][r] : acc0[j][r];
                s += v;  s2 += v * v;
            }
#pragma unroll
            for (int off = 8; off >= 1; off >>= 1) {
                s  += __shfl_xor(s,  off, 32);
                s2 += __shfl_xor(s2, off, 32);
            }
            if (l15 == 0) {
                int rl = wm * 32 + mt * 16 + r + half * 8;
                atomicAdd(&lsum[rl],  s);
                atomicAdd(&lsum2[rl], s2);
            }
        }
    }
    __syncthreads();

#pragma unroll
    for (int mt = 0; mt < 2; ++mt) {
        float mus[8], rss[8];
#pragma unroll
        for (int r = 0; r < 8; ++r) {
            int rl = wm * 32 + mt * 16 + r + half * 8;
            float mu  = lsum[rl]  * (1.f / 512.f);
            float var = lsum2[rl] * (1.f / 512.f) - mu * mu;
            mus[r] = mu;
            rss[r] = rsqrtf(var + 1e-5f);
        }
#pragma unroll
        for (int j = 0; j < 8; ++j) {
            int col = wn * 128 + j * 16 + l15;
            float gg = gam[col], bb = bet[col];
#pragma unroll
            for (int r = 0; r < 8; ++r) {
                int rl = wm * 32 + mt * 16 + r + half * 8;
                float v = mt ? acc1[j][r] : acc0[j][r];
                float o = (v - mus[r]) * rss[r] * gg + bb;
                size_t off = (size_t)(row0 + rl) * 512 + col;
                if (HAS_BASE) o += OutF[off];
                OutF[off] = o;
                OutB[off] = f2bf(o);
            }
        }
    }
}

// ---------------------------------------------------------------------------
// Host orchestration
// ---------------------------------------------------------------------------
extern "C" void kernel_launch(void* const* d_in, const int* in_sizes, int n_in,
                              void* d_out, int out_size, void* d_ws, size_t ws_size,
                              hipStream_t stream) {
    (void)n_in; (void)out_size; (void)ws_size;
    const bool sorted = (in_sizes[0] != N_NODES * DFEAT);
    auto P = [&](int ins, int srt) -> const void* { return d_in[sorted ? srt : ins]; };

    const float* mesh_nfeat = (const float*)P(0, 20);
    const int*   edge_index = (const int*)  P(1, 12);
    const float* mesh_efeat = (const float*)P(2, 19);
    const float* emb_w0 = (const float*)P(3, 17);
    const float* emb_b0 = (const float*)P(4, 13);
    const float* emb_w1 = (const float*)P(5, 18);
    const float* emb_b1 = (const float*)P(6, 14);
    const float* emb_g  = (const float*)P(7, 16);
    const float* emb_be = (const float*)P(8, 15);
    const float* ew0 = (const float*)P(9, 4);
    const float* eb0 = (const float*)P(10, 0);
    const float* ew1 = (const float*)P(11, 5);
    const float* eb1 = (const float*)P(12, 1);
    const float* eg  = (const float*)P(13, 3);
    const float* ebe = (const float*)P(14, 2);
    const float* nw0 = (const float*)P(15, 10);
    const float* nb0 = (const float*)P(16, 6);
    const float* nw1 = (const float*)P(17, 11);
    const float* nb1 = (const float*)P(18, 7);
    const float* ng  = (const float*)P(19, 9);
    const float* nbe = (const float*)P(20, 8);

    const int* srcI = edge_index;
    const int* dstI = edge_index + N_EDGES;

    // -------- workspace carving --------
    char* ws = (char*)d_ws;
    size_t off = 0;
    auto carve = [&](size_t bytes) { void* p = ws + off; off += (bytes + 255) & ~(size_t)255; return p; };
    float*          n_f32  = (float*)         carve((size_t)N_NODES * DFEAT * 4);
    float*          e_f32  = (float*)         carve((size_t)N_EDGES * DFEAT * 4);
    unsigned short* n_bf   = (unsigned short*)carve((size_t)N_NODES * DFEAT * 2);
    unsigned short* e_bf   = (unsigned short*)carve((size_t)N_EDGES * DFEAT * 2);
    unsigned short* agg_bf = (unsigned short*)carve((size_t)N_NODES * DFEAT * 2);
    unsigned short* h_bf   = (unsigned short*)carve((size_t)N_EDGES * DFEAT * 2);
    unsigned int*   w_emb1 = (unsigned int*)  carve((size_t)512  * 512 / 2 * 4);
    unsigned int*   w_ew0  = (unsigned int*)  carve((size_t)NUM_BLOCKS * 1536 * 512 / 2 * 4);
    unsigned int*   w_ew1  = (unsigned int*)  carve((size_t)NUM_BLOCKS * 512  * 512 / 2 * 4);
    unsigned int*   w_nw0  = (unsigned int*)  carve((size_t)NUM_BLOCKS * 1024 * 512 / 2 * 4);
    unsigned int*   w_nw1  = (unsigned int*)  carve((size_t)NUM_BLOCKS * 512  * 512 / 2 * 4);
    int*            deg    = (int*)           carve((size_t)N_NODES * 4);
    int*            offs   = (int*)           carve((size_t)(N_NODES + 1) * 4);
    int*            elist  = (int*)           carve((size_t)N_EDGES * 4);

    // -------- one-time prep (per call; deterministic) --------
    pack_weights_kernel<<<512,   256, 0, stream>>>(emb_w1, w_emb1, 512,  131072);
    pack_weights_kernel<<<24576, 256, 0, stream>>>(ew0,    w_ew0,  1536, 6291456);
    pack_weights_kernel<<<8192,  256, 0, stream>>>(ew1,    w_ew1,  512,  2097152);
    pack_weights_kernel<<<16384, 256, 0, stream>>>(nw0,    w_nw0,  1024, 4194304);
    pack_weights_kernel<<<8192,  256, 0, stream>>>(nw1,    w_nw1,  512,  2097152);
    copyconv_kernel<<<8192, 256, 0, stream>>>(mesh_nfeat, n_f32, n_bf, N_NODES * DFEAT);

    // stable CSR of edges by destination node (deterministic segment-sum order)
    hipMemsetAsync(deg, 0, (size_t)N_NODES * 4, stream);
    degree_kernel<<<N_EDGES / 256, 256, 0, stream>>>(dstI, deg);
    scan_offsets_kernel<<<1, 1024, 0, stream>>>(deg, offs);
    bucket_kernel<<<N_NODES / 256, 256, 0, stream>>>(dstI, offs, elist);

    // -------- edge embedder: e = MLP_LN(mesh_efeat) --------
    embed_h_kernel<<<32768, 256, 0, stream>>>(mesh_efeat, emb_w0, emb_b0, h_bf);
    gemm_ln_kernel<false><<<N_EDGES / 64, 256, 0, stream>>>(
        h_bf, w_emb1, emb_b1, emb_g, emb_be, e_f32, e_bf);

    // -------- 16 interaction blocks --------
    for (int blk = 0; blk < NUM_BLOCKS; ++blk) {
        const unsigned int* wE0 = w_ew0 + (size_t)blk * (1536 * 512 / 2);
        const unsigned int* wE1 = w_ew1 + (size_t)blk * (512  * 512 / 2);
        const unsigned int* wN0 = w_nw0 + (size_t)blk * (1024 * 512 / 2);
        const unsigned int* wN1 = w_nw1 + (size_t)blk * (512  * 512 / 2);

        // edge update: e += LN(MLP(concat(e, n[src], n[dst])))
        gemm_silu_kernel<1536, 2><<<N_EDGES / 64, 256, 0, stream>>>(
            e_bf, n_bf, srcI, dstI, wE0, eb0 + blk * 512, h_bf);
        gemm_ln_kernel<true><<<N_EDGES / 64, 256, 0, stream>>>(
            h_bf, wE1, eb1 + blk * 512, eg + blk * 512, ebe + blk * 512, e_f32, e_bf);

        // node update: n += LN(MLP(concat(n, segment_sum(e, dst))))
        aggregate_gather_kernel<<<N_NODES, 256, 0, stream>>>(e_f32, offs, elist, agg_bf);
        gemm_silu_kernel<1024, 1><<<N_NODES / 64, 256, 0, stream>>>(
            n_bf, agg_bf, nullptr, nullptr, wN0, nb0 + blk * 512, h_bf);
        gemm_ln_kernel<true><<<N_NODES / 64, 256, 0, stream>>>(
            h_bf, wN1, nb1 + blk * 512, ng + blk * 512, nbe + blk * 512, n_f32, n_bf);
    }

    // -------- outputs: (n, e) concatenated --------
    float* out = (float*)d_out;
    hipMemcpyAsync(out, n_f32, (size_t)N_NODES * DFEAT * 4, hipMemcpyDeviceToDevice, stream);
    hipMemcpyAsync(out + (size_t)N_NODES * DFEAT, e_f32, (size_t)N_EDGES * DFEAT * 4,
                   hipMemcpyDeviceToDevice, stream);
}
